// GINEEncoderPPwSkipCat_14697378087542
// MI455X (gfx1250) — matrix operand order repaired and checked by reading, not compiled, not run on hardware
//
#include <hip/hip_runtime.h>
#include <hip/hip_bf16.h>
#include <math.h>

typedef __attribute__((ext_vector_type(16))) _Float16 v16h;
typedef __attribute__((ext_vector_type(8)))  _Float16 v8h;
typedef __attribute__((ext_vector_type(8)))  float    v8f;

#define ACT_LRELU 0
#define ACT_TANH  1

__device__ __forceinline__ float apply_act(float v, int act) {
    if (act == ACT_LRELU) return v > 0.f ? v : 0.01f * v;
    return tanhf(v);
}

// ---------------------------------------------------------------------------
// Async global->LDS copy of 16-byte chunks (CDNA5 GLOBAL_LOAD_ASYNC_TO_LDS_B128,
// tracked by ASYNCcnt; released with s_wait_asynccnt). Generic pointers to LDS
// are {aperture_hi : lds_offset}, and flat->LDS addressing truncates to
// addr[31:0] (ISA 10.2), so the low 32 bits of the shared pointer are the LDS
// address for the VDST operand. nhalves must be a multiple of nthreads*8 so
// every wave executes a uniform trip count around the async ops.
__device__ __forceinline__ void async_copy_b128(const _Float16* __restrict__ gsrc,
                                                _Float16* __restrict__ ldst,
                                                int nhalves, int tid, int nthreads) {
#if defined(__gfx1250__)
    for (int i = tid * 8; i < nhalves; i += nthreads * 8) {
        unsigned lds_off = (unsigned)(unsigned long long)(const void*)(ldst + i);
        unsigned long long ga = (unsigned long long)(const void*)(gsrc + i);
        asm volatile("global_load_async_to_lds_b128 %0, %1, off"
                     :: "v"(lds_off), "v"(ga)
                     : "memory");
    }
    asm volatile("s_wait_asynccnt 0x0" ::: "memory");
#else
    for (int i = tid * 8; i < nhalves; i += nthreads * 8) {
        *(v8h*)(ldst + i) = *(const v8h*)(gsrc + i);
    }
#endif
}

// ---------------------------------------------------------------------------
// WMMA fragment loaders per CDNA5 ISA §7.12.2 layouts.
//
// A (16x32 f16, MxK): lane L holds row M=L&15; hi = L>=16 adds +8 to K base.
//   half j in [0..7]  <-> K = kb + 8*hi + j          (16 contiguous bytes)
//   half j in [8..15] <-> K = kb + 16 + 8*hi + (j-8) (16 contiguous bytes)
// => two ds_load_b128 per fragment from row-major LDS.
__device__ __forceinline__ v16h load_A_frag(const _Float16* __restrict__ A,
                                            int lda, int kb, int lane) {
    int row = lane & 15;
    int hi  = lane >> 4;
    const _Float16* p = A + row * lda + kb + hi * 8;
    v8h lo = *(const v8h*)(p);
    v8h hh = *(const v8h*)(p + 16);
    return __builtin_shufflevector(lo, hh, 0,1,2,3,4,5,6,7,8,9,10,11,12,13,14,15);
}

// B (32x16 f16, KxN): lane L holds column N=n0+(L&15);
//   half j in [0..7] <-> K = kb + 8*hi + j ; half j in [8..15] <-> K = kb + 16 + 8*hi + (j-8)
// Weights kept row-major [out][in] in LDS (B[k][n] = W[n][k]) => two b128 loads.
__device__ __forceinline__ v16h load_B_frag(const _Float16* __restrict__ W,
                                            int ldw, int n0, int kb, int lane) {
    int n  = n0 + (lane & 15);
    int hi = lane >> 4;
    const _Float16* p = W + n * ldw + kb + hi * 8;
    v8h lo = *(const v8h*)(p);
    v8h hh = *(const v8h*)(p + 16);
    return __builtin_shufflevector(lo, hh, 0,1,2,3,4,5,6,7,8,9,10,11,12,13,14,15);
}

// One 16-row GEMM: D[16,Nout] = act(A[16,K] * W^T + bias), result -> LDS f16.
__device__ __forceinline__ void gemm_lds(const _Float16* __restrict__ A, int lda,
                                         const _Float16* __restrict__ W, int ldw,
                                         const float* __restrict__ bias,
                                         _Float16* __restrict__ D, int ldd,
                                         int Nout, int K, int lane, int act) {
    int nloc = lane & 15, hi = lane >> 4;
    for (int n0 = 0; n0 < Nout; n0 += 16) {
        v8f c = {};
        for (int kb = 0; kb < K; kb += 32) {
            v16h a = load_A_frag(A, lda, kb, lane);
            v16h b = load_B_frag(W, ldw, n0, kb, lane);
            c = __builtin_amdgcn_wmma_f32_16x16x32_f16(false, a, false, b,
                                                       (short)0, c, false, false);
        }
        float bv = bias[n0 + nloc];
#pragma unroll
        for (int i = 0; i < 8; i++) {
            float v = apply_act(c[i] + bv, act);
            D[(i + 8 * hi) * ldd + n0 + nloc] = (_Float16)v;  // C/D: VGPR i -> M=i+8*hi
        }
    }
}

// Same but final result -> global f32.
__device__ __forceinline__ void gemm_glb(const _Float16* __restrict__ A, int lda,
                                         const _Float16* __restrict__ W, int ldw,
                                         const float* __restrict__ bias,
                                         float* __restrict__ D, int ldd,
                                         int Nout, int K, int lane, int act) {
    int nloc = lane & 15, hi = lane >> 4;
    for (int n0 = 0; n0 < Nout; n0 += 16) {
        v8f c = {};
        for (int kb = 0; kb < K; kb += 32) {
            v16h a = load_A_frag(A, lda, kb, lane);
            v16h b = load_B_frag(W, ldw, n0, kb, lane);
            c = __builtin_amdgcn_wmma_f32_16x16x32_f16(false, a, false, b,
                                                       (short)0, c, false, false);
        }
        float bv = bias[n0 + nloc];
#pragma unroll
        for (int i = 0; i < 8; i++) {
            float v = apply_act(c[i] + bv, act);
            D[(long long)(i + 8 * hi) * ldd + n0 + nloc] = v;
        }
    }
}

// ---------------------------------------------------------------------------
// Packed f16 weight layout inside workspace (offsets in halves):
//   prep w0 padded [128][32] @ 0        (4096)
//   prep w1        [128][128] @ 4096    (16384)
//   mlp  w0/w1/w2  [128][128] @ 20480/36864/53248
//   post w0        [128][256] @ 69632   (32768)
//   post w1        [64][128]  @ 102400  (8192)   -> total 110592 halves
#define PK_W0P   0
#define PK_W1    4096
#define PK_MW0   20480
#define PK_TOTAL 110592

// Kernel P: one-time f32 -> f16 weight pack (runs every launch; deterministic).
__global__ void pack_kernel(const float* __restrict__ w0, const float* __restrict__ w1,
                            const float* __restrict__ mw0, const float* __restrict__ mw1,
                            const float* __restrict__ mw2, const float* __restrict__ pw0,
                            const float* __restrict__ pw1, _Float16* __restrict__ pk) {
    int stride = gridDim.x * blockDim.x;
    int t = blockIdx.x * blockDim.x + threadIdx.x;
    for (int i = t; i < 128 * 16; i += stride) {
        int o = i >> 4, c = i & 15;
        pk[PK_W0P + o * 32 + c]      = (_Float16)w0[i];
        pk[PK_W0P + o * 32 + 16 + c] = (_Float16)0.f;     // K padding 16->32
    }
    for (int i = t; i < 16384; i += stride) {
        pk[PK_W1 + i]          = (_Float16)w1[i];
        pk[PK_MW0 + i]         = (_Float16)mw0[i];
        pk[PK_MW0 + 16384 + i] = (_Float16)mw1[i];
        pk[PK_MW0 + 32768 + i] = (_Float16)mw2[i];
    }
    for (int i = t; i < 32768; i += stride) pk[PK_MW0 + 49152 + i] = (_Float16)pw0[i];
    for (int i = t; i < 8192; i += stride)  pk[PK_MW0 + 81920 + i] = (_Float16)pw1[i];
}

// ---------------------------------------------------------------------------
// Kernel 0: zero the aggregation buffer (scatter target).
__global__ void zero_kernel(float* __restrict__ p, long long n) {
    long long i = (long long)blockIdx.x * blockDim.x + threadIdx.x;
    if (i < n) p[i] = 0.f;
}

// ---------------------------------------------------------------------------
// Kernel 1: prep MLP  h = tanh(lrelu(x*W0^T+b0)*W1^T+b1)
// 256 threads = 8 waves, each wave owns one 16-node tile (128 nodes / block).
// Dynamic LDS (bytes): 0: w0p+w1 f16 (contiguous, 20480 halves = 40960 B)
//                      40960: b0s/b1s f32 (1024 B)
//                      41984: per-wave xt[16][32]+h1[16][128] = 2560 halves * 8
#define K1_SMEM_BYTES 82944
__global__ void prep_kernel(const float* __restrict__ x,
                            const _Float16* __restrict__ pk,
                            const float* __restrict__ b0, const float* __restrict__ b1,
                            float* __restrict__ h, int nNodes) {
    extern __shared__ __align__(16) char smem[];
    _Float16* ws   = (_Float16*)(smem);          // w0p @0, w1 @4096 halves
    float*    b0s  = (float*)(smem + 40960);
    float*    b1s  = (float*)(smem + 41472);
    _Float16* actb = (_Float16*)(smem + 41984);

    int tid = threadIdx.x;
    int wave = tid >> 5, lane = tid & 31;
    int tile = blockIdx.x * 8 + wave;
    int base = tile * 16;

    // prefetch this wave's x tile while weights stream into LDS
    if (base < nNodes) {
        for (int r = lane; r < 16; r += 32)
            __builtin_prefetch(x + (long long)(base + r) * 16, 0, 3);
    }

    async_copy_b128(pk, ws, 20480, tid, 256);    // w0p + w1, one contiguous DMA
    for (int i = tid; i < 128; i += 256) { b0s[i] = b0[i]; b1s[i] = b1[i]; }
    __syncthreads();

    if (base >= nNodes) return;

    _Float16* xt = actb + wave * 2560;   // [16][32]
    _Float16* h1 = xt + 512;             // [16][128]

    for (int i = lane; i < 16 * 32; i += 32) {
        int r = i >> 5, c = i & 31;
        float v = 0.f;
        if (c < 16 && (base + r) < nNodes) v = x[(long long)(base + r) * 16 + c];
        xt[r * 32 + c] = (_Float16)v;
    }
    // per-wave LDS buffers: same-wave DS ops are in-order, no barrier needed
    gemm_lds(xt, 32, ws, 32, b0s, h1, 128, 128, 32, lane, ACT_LRELU);
    gemm_glb(h1, 128, ws + PK_W1, 128, b1s, h + (long long)base * 128, 128,
             128, 128, lane, ACT_TANH);
}

// ---------------------------------------------------------------------------
// Kernel 2: GINE message + scatter.
//   e[d]  = ew * edge_w[d] + edge_b[d]  (rank-1 edge lin, EDGE_DIM=1)
//   m     = relu(h[src] + e) ; agg[dst] += m  (f32 atomics)
// One thread = (edge, 4 channels). Pure HBM/L2-atomic bound phase.
__global__ void edge_kernel(const long long* __restrict__ ei,
                            const float* __restrict__ ew,
                            const float* __restrict__ ecw,
                            const float* __restrict__ ecb,
                            const float* __restrict__ h,
                            float* __restrict__ agg, int nEdges) {
    long long idx = (long long)blockIdx.x * blockDim.x + threadIdx.x;
    int e = (int)(idx >> 5);
    if (e >= nEdges) return;
    int d = ((int)idx & 31) * 4;

    int src = (int)ei[e];
    int dst = (int)ei[nEdges + e];
    float w = ew[e];

    const float4 hv = *(const float4*)(h + (long long)src * 128 + d);
    const float4 cw = *(const float4*)(ecw + d);
    const float4 cb = *(const float4*)(ecb + d);

    float m0 = hv.x + w * cw.x + cb.x;
    float m1 = hv.y + w * cw.y + cb.y;
    float m2 = hv.z + w * cw.z + cb.z;
    float m3 = hv.w + w * cw.w + cb.w;

    float* ap = agg + (long long)dst * 128 + d;
    if (m0 > 0.f) atomicAdd(ap + 0, m0);   // relu: adding 0 is a no-op, skip
    if (m1 > 0.f) atomicAdd(ap + 1, m1);
    if (m2 > 0.f) atomicAdd(ap + 2, m2);
    if (m3 > 0.f) atomicAdd(ap + 3, m3);
}

// ---------------------------------------------------------------------------
// Kernel 3: fused  z=h+agg -> mlp0 -> mlp1 -> mlp2(tanh) -> cat(z,r) -> post0
//           -> post1(tanh) -> out[N,64].  All weights resident in LDS (f16).
// 128 threads = 4 waves, each wave one 16-node tile.
// Dynamic LDS (bytes):
//   0      : mlp w0/w1/w2 + post w0 + post w1 f16 (contiguous, 90112 halves)
//   180224 : biases f32 (128*4 + 64 = 576)
//   182528 : per-wave act: ZR[16][256] + T1[16][128] + T2[16][128] = 8192 halves
#define K3_SMEM_BYTES 248064
__global__ void mlp_post_kernel(const float* __restrict__ h, const float* __restrict__ agg,
                                const _Float16* __restrict__ pk,
                                const float* __restrict__ b0, const float* __restrict__ b1,
                                const float* __restrict__ b2, const float* __restrict__ pb0,
                                const float* __restrict__ pb1,
                                float* __restrict__ out, int nNodes) {
    extern __shared__ __align__(16) char smem[];
    _Float16* w0s  = (_Float16*)(smem);     // [128][128]
    _Float16* w1s  = w0s + 16384;           // [128][128]
    _Float16* w2s  = w1s + 16384;           // [128][128]
    _Float16* pw0s = w2s + 16384;           // [128][256]
    _Float16* pw1s = pw0s + 32768;          // [64][128]
    float*    bss  = (float*)(pw1s + 8192);
    float *b0s = bss, *b1s = bss + 128, *b2s = bss + 256, *pb0s = bss + 384,
          *pb1s = bss + 512;
    _Float16* actb = (_Float16*)(bss + 576);

    int tid = threadIdx.x;
    int wave = tid >> 5, lane = tid & 31;
    int tile = blockIdx.x * 4 + wave;
    int base = tile * 16;

    // prefetch this wave's node tile while weights stream into LDS
    if (base < nNodes) {
        for (int r = lane; r < 16; r += 32) {
            __builtin_prefetch(h + (long long)(base + r) * 128, 0, 3);
            __builtin_prefetch(agg + (long long)(base + r) * 128, 0, 3);
        }
    }

    // all five weight matrices are contiguous in the packed buffer and in LDS:
    // one 176 KB async DMA (global -> LDS), no VGPR roundtrip, no convert
    async_copy_b128(pk + PK_MW0, w0s, 90112, tid, 128);
    for (int i = tid; i < 128; i += 128) {
        b0s[i] = b0[i]; b1s[i] = b1[i]; b2s[i] = b2[i]; pb0s[i] = pb0[i];
    }
    for (int i = tid; i < 64; i += 128) pb1s[i] = pb1[i];
    __syncthreads();

    if (base >= nNodes) return;

    _Float16* ZR = actb + wave * 8192;  // [16][256]: cols 0..127 z_final, 128..255 r
    _Float16* T1 = ZR + 4096;           // [16][128]
    _Float16* T2 = T1 + 2048;           // [16][128]

    for (int i = lane; i < 16 * 128; i += 32) {
        int r = i >> 7, c = i & 127;
        long long g = (long long)(base + r) * 128 + c;
        float hv = h[g];
        T1[r * 128 + c] = (_Float16)(hv + agg[g]);   // z = h + agg
        ZR[r * 256 + 128 + c] = (_Float16)hv;        // r (skip branch)
    }

    gemm_lds(T1, 128, w0s, 128, b0s, T2, 128, 128, 128, lane, ACT_LRELU);  // mlp0
    gemm_lds(T2, 128, w1s, 128, b1s, T1, 128, 128, 128, lane, ACT_LRELU);  // mlp1
    gemm_lds(T1, 128, w2s, 128, b2s, ZR, 256, 128, 128, lane, ACT_TANH);   // mlp2 -> ZR[:,0:128]
    // post0: K=256 over cat(z, r) — single GEMM since ZR is contiguous [16][256]
    gemm_lds(ZR, 256, pw0s, 256, pb0s, T1, 128, 128, 256, lane, ACT_LRELU);
    // post1 -> out [16,64], tanh
    gemm_glb(T1, 128, pw1s, 128, pb1s, out + (long long)base * 64, 64,
             64, 128, lane, ACT_TANH);
}

// ---------------------------------------------------------------------------
extern "C" void kernel_launch(void* const* d_in, const int* in_sizes, int n_in,
                              void* d_out, int out_size, void* d_ws, size_t ws_size,
                              hipStream_t stream) {
    const float*     x   = (const float*)d_in[0];
    const long long* ei  = (const long long*)d_in[1];   // int64 edge_index [2,E]
    const float*     ew  = (const float*)d_in[2];       // [E,1]
    const float *pw0_ = (const float*)d_in[3],  *pb0_ = (const float*)d_in[4];
    const float *pw1_ = (const float*)d_in[5],  *pb1_ = (const float*)d_in[6];
    const float *ecw  = (const float*)d_in[7],  *ecb  = (const float*)d_in[8];
    const float *mw0  = (const float*)d_in[9],  *mb0  = (const float*)d_in[10];
    const float *mw1  = (const float*)d_in[11], *mb1  = (const float*)d_in[12];
    const float *mw2  = (const float*)d_in[13], *mb2  = (const float*)d_in[14];
    const float *qw0  = (const float*)d_in[15], *qb0  = (const float*)d_in[16];
    const float *qw1  = (const float*)d_in[17], *qb1  = (const float*)d_in[18];
    float* out = (float*)d_out;

    int N = in_sizes[0] / 16;   // 100000
    int E = in_sizes[2];        // 400000 (EDGE_DIM=1)

    float*     h   = (float*)d_ws;                    // [N,128]
    float*     agg = h + (size_t)N * 128;             // [N,128]
    _Float16*  pk  = (_Float16*)(agg + (size_t)N * 128);  // packed f16 weights

    // allow >64KB dynamic LDS (host-side attribute, capture-safe)
    (void)hipFuncSetAttribute((const void*)prep_kernel,
                              hipFuncAttributeMaxDynamicSharedMemorySize, K1_SMEM_BYTES);
    (void)hipFuncSetAttribute((const void*)mlp_post_kernel,
                              hipFuncAttributeMaxDynamicSharedMemorySize, K3_SMEM_BYTES);

    long long nz = (long long)N * 128;
    zero_kernel<<<(unsigned)((nz + 255) / 256), 256, 0, stream>>>(agg, nz);

    pack_kernel<<<128, 256, 0, stream>>>(pw0_, pw1_, mw0, mw1, mw2, qw0, qw1, pk);

    int ntiles = (N + 15) / 16;
    prep_kernel<<<(ntiles + 7) / 8, 256, K1_SMEM_BYTES, stream>>>(
        x, pk, pb0_, pb1_, h, N);

    long long ethreads = (long long)E * 32;
    edge_kernel<<<(unsigned)((ethreads + 255) / 256), 256, 0, stream>>>(
        ei, ew, ecw, ecb, h, agg, E);

    mlp_post_kernel<<<(ntiles + 3) / 4, 128, K3_SMEM_BYTES, stream>>>(
        h, agg, pk, mb0, mb1, mb2, qb0, qb1, out, N);
}
